// Block_8040178778841
// MI455X (gfx1250) — matrix operand order repaired and checked by reading, not compiled, not run on hardware
//
#include <hip/hip_runtime.h>
#include <stdint.h>

// ---------------------------------------------------------------------------
// Types
// ---------------------------------------------------------------------------
typedef __bf16 bf16_t;
typedef bf16_t v16bf __attribute__((ext_vector_type(16)));
typedef bf16_t v8bf  __attribute__((ext_vector_type(8)));
typedef float  v8f   __attribute__((ext_vector_type(8)));

static constexpr int Bq  = 2;
static constexpr int Tq  = 4096;
static constexpr int Cq  = 512;
static constexpr int Hq  = 8;
static constexpr int Dq  = 64;
static constexpr int FFq = 2048;
static constexpr int Rq  = Bq * Tq;          // 8192 rows of activations

__device__ __forceinline__ bf16_t f2bf(float f) {
    uint32_t u = __builtin_bit_cast(uint32_t, f);
    uint32_t r = u + 0x7FFFu + ((u >> 16) & 1u);     // round-to-nearest-even
    uint16_t h = (uint16_t)(r >> 16);
    return __builtin_bit_cast(bf16_t, h);
}

__device__ __forceinline__ v16bf ld_frag16(const bf16_t* p0, const bf16_t* p1) {
    v8bf lo = *(const v8bf*)p0;
    v8bf hi = *(const v8bf*)p1;
    return __builtin_shufflevector(lo, hi, 0,1,2,3,4,5,6,7,8,9,10,11,12,13,14,15);
}

__device__ __forceinline__ v8f wmma_bf16(v16bf a, v16bf b, v8f c) {
    // D = A(16x32 bf16) * B(32x16 bf16) + C(16x16 f32)
    return __builtin_amdgcn_wmma_f32_16x16x32_bf16(
        false, a, false, b, (short)0, c, false, false);
}

// LDS byte address of a __shared__ object: generic shared pointers are
// {shared_aperture_hi32, lds_offset} (ISA 10.2 aperture rules), so the low
// 32 bits are the LDS address the async/DS units consume.
__device__ __forceinline__ uint32_t lds_addr32(const void* p) {
    return (uint32_t)(uintptr_t)p;
}

// Async global -> LDS copy, 16B per lane, tracked by ASYNCcnt.
#define ASYNC_COPY_B128(dst_lds_u32, src_ptr)                                  \
    asm volatile("global_load_async_to_lds_b128 %0, %1, off"                   \
                 :: "v"(dst_lds_u32),                                          \
                    "v"((unsigned long long)(uintptr_t)(src_ptr))              \
                 : "memory")

#define WAIT_ASYNCCNT0()                                                       \
    asm volatile("s_wait_asynccnt 0x0" ::: "memory")

// ---------------------------------------------------------------------------
// Weight convert + transpose: in f32 [K,N] -> out bf16 [N,K]
// ---------------------------------------------------------------------------
__global__ void convert_transpose_kernel(const float* __restrict__ in,
                                         bf16_t* __restrict__ out,
                                         int K, int N) {
    int idx = blockIdx.x * 256 + threadIdx.x;
    if (idx >= K * N) return;
    int k = idx / N;
    int n = idx - k * N;
    out[(size_t)n * K + k] = f2bf(in[idx]);
}

// ---------------------------------------------------------------------------
// LayerNorm over rows of 512, output bf16. One block (256 threads) per row.
// ---------------------------------------------------------------------------
__global__ __launch_bounds__(256)
void ln_kernel(const float* __restrict__ x, const float* __restrict__ g,
               const float* __restrict__ bt, bf16_t* __restrict__ out) {
    int row  = blockIdx.x;
    int tid  = threadIdx.x;
    const float* xr = x + (size_t)row * Cq;
    float v0 = xr[tid], v1 = xr[tid + 256];
    float s = v0 + v1;
    float q = v0 * v0 + v1 * v1;
#pragma unroll
    for (int m = 16; m >= 1; m >>= 1) {
        s += __shfl_xor(s, m);
        q += __shfl_xor(q, m);
    }
    __shared__ float rs[8], rq[8];
    int w = tid >> 5, lane = tid & 31;
    if (lane == 0) { rs[w] = s; rq[w] = q; }
    __syncthreads();
    if (tid == 0) {
        float ts = 0.f, tq = 0.f;
#pragma unroll
        for (int i = 0; i < 8; ++i) { ts += rs[i]; tq += rq[i]; }
        rs[0] = ts; rq[0] = tq;
    }
    __syncthreads();
    float mu   = rs[0] * (1.0f / Cq);
    float var  = rq[0] * (1.0f / Cq) - mu * mu;
    float rstd = rsqrtf(var + 1e-5f);
    bf16_t* orow = out + (size_t)row * Cq;
    orow[tid]       = f2bf((v0 - mu) * rstd * g[tid]       + bt[tid]);
    orow[tid + 256] = f2bf((v1 - mu) * rstd * g[tid + 256] + bt[tid + 256]);
}

// ---------------------------------------------------------------------------
// bf16 WMMA GEMM:  out[M,N] = A[M,K] * Bt[N,K]^T  (+bias) (+relu) (+residual)
// Workgroup: 256 threads = 8 waves (2x4), tile 64x128; wave tile 32x32.
// K-loop is double-buffered: tiles staged with global_load_async_to_lds_b128
// (ASYNCcnt) and the next tile is prefetched while WMMAs run on the current.
// ---------------------------------------------------------------------------
template <bool BIAS, bool RELU, bool RESID>
__global__ __launch_bounds__(256)
void gemm_bf16_kernel(const bf16_t* __restrict__ A,
                      const bf16_t* __restrict__ Bt,
                      const float* __restrict__ bias,
                      const float* __restrict__ resid,
                      void* __restrict__ outp,
                      int M, int N, int K) {
    __shared__ __align__(16) bf16_t As[2][64 * 32];
    __shared__ __align__(16) bf16_t Bs[2][128 * 32];

    const int tid   = threadIdx.x;
    const int lane  = tid & 31;
    const int w     = tid >> 5;
    const int wr    = w >> 2;        // 0..1
    const int wc    = w & 3;         // 0..3
    const int m0    = blockIdx.y * 64;
    const int n0    = blockIdx.x * 128;
    const int khalf = lane >> 4;     // 0..1
    const int rsel  = lane & 15;     // 0..15

    // cooperative staging coordinates
    const int arow = tid >> 2;               // 0..63
    const int acol = (tid & 3) * 8;          // 0,8,16,24

    auto stage_tiles = [&](int buf, int kk) {
        // A tile 64x32: one async b128 per thread
        ASYNC_COPY_B128(lds_addr32(&As[buf][arow * 32 + acol]),
                        &A[(size_t)(m0 + arow) * K + kk + acol]);
        // B tile 128x32: two async b128 per thread
#pragma unroll
        for (int c = tid; c < 512; c += 256) {
            int row = c >> 2;
            int col = (c & 3) * 8;
            ASYNC_COPY_B128(lds_addr32(&Bs[buf][row * 32 + col]),
                            &Bt[(size_t)(n0 + row) * K + kk + col]);
        }
    };

    v8f acc[2][2] = {};

    stage_tiles(0, 0);

    for (int kk = 0; kk < K; kk += 32) {
        const int cur = (kk >> 5) & 1;
        WAIT_ASYNCCNT0();
        __syncthreads();                       // tile `cur` visible to all waves
        if (kk + 32 < K) {
            stage_tiles(1 - cur, kk + 32);     // overlap copy with compute
            if (kk + 64 < K)                   // pull next-next lines into GL2
                __builtin_prefetch(&A[(size_t)(m0 + arow) * K + kk + 64 + acol],
                                   0, 0);
        }

        v16bf af[2], bfv[2];
#pragma unroll
        for (int fr = 0; fr < 2; ++fr) {
            int row = wr * 32 + fr * 16 + rsel;
            af[fr] = ld_frag16(&As[cur][row * 32 + khalf * 8],
                               &As[cur][row * 32 + 16 + khalf * 8]);
        }
#pragma unroll
        for (int fc = 0; fc < 2; ++fc) {
            int nrow = wc * 32 + fc * 16 + rsel;
            bfv[fc] = ld_frag16(&Bs[cur][nrow * 32 + khalf * 8],
                                &Bs[cur][nrow * 32 + 16 + khalf * 8]);
        }
#pragma unroll
        for (int fr = 0; fr < 2; ++fr)
#pragma unroll
            for (int fc = 0; fc < 2; ++fc)
                acc[fr][fc] = wmma_bf16(af[fr], bfv[fc], acc[fr][fc]);
        // no trailing barrier: next iteration's wait+barrier orders the
        // buffer reuse (all waves' reads of `cur` precede that barrier).
    }

    // epilogue
#pragma unroll
    for (int fr = 0; fr < 2; ++fr)
#pragma unroll
        for (int fc = 0; fc < 2; ++fc)
#pragma unroll
            for (int r = 0; r < 8; ++r) {
                int m = m0 + wr * 32 + fr * 16 + khalf * 8 + r;
                int n = n0 + wc * 32 + fc * 16 + rsel;
                float v = acc[fr][fc][r];
                if (BIAS)  v += bias[n];
                if (RELU)  v = fmaxf(v, 0.0f);
                size_t off = (size_t)m * N + n;
                if (RESID) ((float*)outp)[off] = resid[off] + v;
                else       ((bf16_t*)outp)[off] = f2bf(v);
            }
}

// ---------------------------------------------------------------------------
// Flash attention (causal), bf16 WMMA, f32 online softmax.
// Grid: (T/128, B*H). Block: 256 threads = 8 waves; wave = 16 query rows.
// Q/K/V layout: [b*T + t][h*64 + d] bf16 (row stride 512).
// K chunk staged via async-to-LDS; V transposed through registers into LDS.
// ---------------------------------------------------------------------------
__global__ __launch_bounds__(256)
void flash_attn_kernel(const bf16_t* __restrict__ Q,
                       const bf16_t* __restrict__ Kb,
                       const bf16_t* __restrict__ Vb,
                       bf16_t* __restrict__ O,
                       float scale) {
    __shared__ __align__(16) bf16_t Ks[32 * 64];      // [key][d]
    __shared__ __align__(16) bf16_t Vt[64 * 32];      // [d][key]
    __shared__ __align__(16) bf16_t Pl[8 * 16 * 32];  // per-wave P tile

    const int tid   = threadIdx.x;
    const int lane  = tid & 31;
    const int w     = tid >> 5;
    const int khalf = lane >> 4;
    const int rsel  = lane & 15;
    const int bh    = blockIdx.y;
    const int b     = bh >> 3;            // H = 8
    const int h     = bh & 7;
    const int qblk  = blockIdx.x * 128;
    const int q0w   = qblk + w * 16;
    const size_t headoff = (size_t)h * Dq;

    // Q A-fragments (16 rows x 64 K, two 16x32 frags), contiguous 16B loads
    const bf16_t* qrow = Q + ((size_t)(b * Tq + q0w + rsel)) * Cq + headoff;
    v16bf qa[2];
    qa[0] = ld_frag16(qrow + khalf * 8,      qrow + 16 + khalf * 8);
    qa[1] = ld_frag16(qrow + 32 + khalf * 8, qrow + 48 + khalf * 8);

    v8f oacc[4] = {};
    float m_run[8], l_run[8];
#pragma unroll
    for (int r = 0; r < 8; ++r) { m_run[r] = -1e30f; l_run[r] = 0.0f; }

    bf16_t* Pw = &Pl[w * 512];
    const int kend = qblk + 128;
    const int kr   = tid >> 3;            // staging coords
    const int d0   = (tid & 7) * 8;

    for (int kc = 0; kc < kend; kc += 32) {
        // K chunk [32][64]: async global -> LDS (16B per lane)
        const bf16_t* kp =
            Kb + ((size_t)(b * Tq + kc + kr)) * Cq + headoff + d0;
        ASYNC_COPY_B128(lds_addr32(&Ks[kr * 64 + d0]), kp);
        // V chunk transposed -> Vt[64][32] through registers
        const bf16_t* vp =
            Vb + ((size_t)(b * Tq + kc + kr)) * Cq + headoff + d0;
        v8bf vv = *(const v8bf*)vp;
#pragma unroll
        for (int j = 0; j < 8; ++j) Vt[(d0 + j) * 32 + kr] = vv[j];
        // prefetch next chunk's lines into GL2
        if (kc + 32 < kend) {
            __builtin_prefetch(kp + (size_t)32 * Cq, 0, 0);
            __builtin_prefetch(vp + (size_t)32 * Cq, 0, 0);
        }
        WAIT_ASYNCCNT0();
        __syncthreads();

        if (kc <= q0w + 15) {   // wave-uniform causal skip
            // S = Q * K^T : 16x32, two 16x16 C-frags, 4 WMMA
            v8f s[2];
#pragma unroll
            for (int nf = 0; nf < 2; ++nf) {
                int nl = nf * 16 + rsel;
                v16bf kb0 = ld_frag16(&Ks[nl * 64 + khalf * 8],
                                      &Ks[nl * 64 + 16 + khalf * 8]);
                v16bf kb1 = ld_frag16(&Ks[nl * 64 + 32 + khalf * 8],
                                      &Ks[nl * 64 + 48 + khalf * 8]);
                v8f sa = {};
                sa = wmma_bf16(qa[0], kb0, sa);
                sa = wmma_bf16(qa[1], kb1, sa);
                s[nf] = sa;
            }
            // scale + causal mask + online softmax
#pragma unroll
            for (int r = 0; r < 8; ++r) {
                int qi = q0w + khalf * 8 + r;
                float s0 = s[0][r] * scale;
                float s1 = s[1][r] * scale;
                if (kc + rsel > qi)      s0 = -1e30f;
                if (kc + 16 + rsel > qi) s1 = -1e30f;
                float mx = fmaxf(s0, s1);
#pragma unroll
                for (int msk = 1; msk <= 8; msk <<= 1)
                    mx = fmaxf(mx, __shfl_xor(mx, msk));
                float mnew  = fmaxf(m_run[r], mx);
                float alpha = __expf(m_run[r] - mnew);
                m_run[r] = mnew;
                float p0 = __expf(s0 - mnew);
                float p1 = __expf(s1 - mnew);
                float rsum = p0 + p1;
#pragma unroll
                for (int msk = 1; msk <= 8; msk <<= 1)
                    rsum += __shfl_xor(rsum, msk);
                l_run[r] = l_run[r] * alpha + rsum;
#pragma unroll
                for (int df = 0; df < 4; ++df) oacc[df][r] *= alpha;
                Pw[(khalf * 8 + r) * 32 + rsel]      = f2bf(p0);
                Pw[(khalf * 8 + r) * 32 + 16 + rsel] = f2bf(p1);
            }
            // P as A-fragment (same-wave LDS round trip; DS ops in-order)
            v16bf pa = ld_frag16(&Pw[rsel * 32 + khalf * 8],
                                 &Pw[rsel * 32 + 16 + khalf * 8]);
            // O += P * V : 4 WMMA over d tiles
#pragma unroll
            for (int df = 0; df < 4; ++df) {
                int dl = df * 16 + rsel;
                v16bf vb = ld_frag16(&Vt[dl * 32 + khalf * 8],
                                     &Vt[dl * 32 + 16 + khalf * 8]);
                oacc[df] = wmma_bf16(pa, vb, oacc[df]);
            }
        }
        __syncthreads();
    }

    // normalize and store attn output (bf16)
#pragma unroll
    for (int df = 0; df < 4; ++df)
#pragma unroll
        for (int r = 0; r < 8; ++r) {
            int qi = q0w + khalf * 8 + r;
            float val = oacc[df][r] / l_run[r];
            O[((size_t)(b * Tq + qi)) * Cq + headoff + df * 16 + rsel] =
                f2bf(val);
        }
}

// ---------------------------------------------------------------------------
// Host-side launch
// ---------------------------------------------------------------------------
extern "C" void kernel_launch(void* const* d_in, const int* in_sizes, int n_in,
                              void* d_out, int out_size, void* d_ws,
                              size_t ws_size, hipStream_t stream) {
    (void)in_sizes; (void)n_in; (void)out_size; (void)ws_size;
    const float* x   = (const float*)d_in[0];
    const float* Wq  = (const float*)d_in[1];
    const float* Wk  = (const float*)d_in[2];
    const float* Wv  = (const float*)d_in[3];
    const float* Wo  = (const float*)d_in[4];
    const float* bo  = (const float*)d_in[5];
    const float* W1  = (const float*)d_in[6];
    const float* b1  = (const float*)d_in[7];
    const float* W2  = (const float*)d_in[8];
    const float* b2  = (const float*)d_in[9];
    const float* g1  = (const float*)d_in[10];
    const float* be1 = (const float*)d_in[11];
    const float* g2  = (const float*)d_in[12];
    const float* be2 = (const float*)d_in[13];
    float* out = (float*)d_out;

    // workspace carve (256B aligned)
    char* p = (char*)d_ws;
    auto alloc = [&](size_t bytes) {
        void* r = (void*)p;
        p += (bytes + 255) & ~(size_t)255;
        return r;
    };
    bf16_t* WqT = (bf16_t*)alloc((size_t)Cq * Cq * 2);
    bf16_t* WkT = (bf16_t*)alloc((size_t)Cq * Cq * 2);
    bf16_t* WvT = (bf16_t*)alloc((size_t)Cq * Cq * 2);
    bf16_t* WoT = (bf16_t*)alloc((size_t)Cq * Cq * 2);
    bf16_t* W1T = (bf16_t*)alloc((size_t)Cq * FFq * 2);
    bf16_t* W2T = (bf16_t*)alloc((size_t)FFq * Cq * 2);
    bf16_t* h1  = (bf16_t*)alloc((size_t)Rq * Cq * 2);
    bf16_t* qb  = (bf16_t*)alloc((size_t)Rq * Cq * 2);
    bf16_t* kb  = (bf16_t*)alloc((size_t)Rq * Cq * 2);
    bf16_t* vb  = (bf16_t*)alloc((size_t)Rq * Cq * 2);
    bf16_t* at  = (bf16_t*)alloc((size_t)Rq * Cq * 2);
    float*  x1  = (float*) alloc((size_t)Rq * Cq * 4);
    bf16_t* h2  = (bf16_t*)alloc((size_t)Rq * Cq * 2);
    bf16_t* ff  = (bf16_t*)alloc((size_t)Rq * FFq * 2);

    // 1) weight convert + transpose to bf16 [N][K]
    {
        int n1 = Cq * Cq;
        convert_transpose_kernel<<<n1 / 256, 256, 0, stream>>>(Wq, WqT, Cq, Cq);
        convert_transpose_kernel<<<n1 / 256, 256, 0, stream>>>(Wk, WkT, Cq, Cq);
        convert_transpose_kernel<<<n1 / 256, 256, 0, stream>>>(Wv, WvT, Cq, Cq);
        convert_transpose_kernel<<<n1 / 256, 256, 0, stream>>>(Wo, WoT, Cq, Cq);
        int n2 = Cq * FFq;
        convert_transpose_kernel<<<n2 / 256, 256, 0, stream>>>(W1, W1T, Cq, FFq);
        convert_transpose_kernel<<<n2 / 256, 256, 0, stream>>>(W2, W2T, FFq, Cq);
    }

    // 2) LN1 -> bf16
    ln_kernel<<<Rq, 256, 0, stream>>>(x, g1, be1, h1);

    // 3) Q/K/V projections (no bias)
    dim3 gqkv(Cq / 128, Rq / 64);
    gemm_bf16_kernel<false, false, false><<<gqkv, 256, 0, stream>>>(
        h1, WqT, nullptr, nullptr, qb, Rq, Cq, Cq);
    gemm_bf16_kernel<false, false, false><<<gqkv, 256, 0, stream>>>(
        h1, WkT, nullptr, nullptr, kb, Rq, Cq, Cq);
    gemm_bf16_kernel<false, false, false><<<gqkv, 256, 0, stream>>>(
        h1, WvT, nullptr, nullptr, vb, Rq, Cq, Cq);

    // 4) causal flash attention (scale = C^-0.5 per reference)
    flash_attn_kernel<<<dim3(Tq / 128, Bq * Hq), 256, 0, stream>>>(
        qb, kb, vb, at, 0.044194173824159216f);

    // 5) output projection + bias + residual -> x1 (f32)
    gemm_bf16_kernel<true, false, true><<<gqkv, 256, 0, stream>>>(
        at, WoT, bo, x, x1, Rq, Cq, Cq);

    // 6) LN2 -> bf16
    ln_kernel<<<Rq, 256, 0, stream>>>(x1, g2, be2, h2);

    // 7) FFN up + bias + relu -> bf16
    gemm_bf16_kernel<true, true, false><<<dim3(FFq / 128, Rq / 64), 256, 0,
                                          stream>>>(
        h2, W1T, b1, nullptr, ff, Rq, FFq, Cq);

    // 8) FFN down + bias + residual -> d_out (f32)
    gemm_bf16_kernel<true, false, true><<<gqkv, 256, 0, stream>>>(
        ff, W2T, b2, x1, out, Rq, Cq, FFq);
}
